// ODERNNCell_65111704207712
// MI455X (gfx1250) — matrix operand order repaired and checked by reading, not compiled, not run on hardware
//
#include <hip/hip_runtime.h>

typedef __bf16 bf16_t;
typedef __attribute__((ext_vector_type(16))) bf16_t v16bf;
typedef __attribute__((ext_vector_type(4)))  bf16_t v4bf;
typedef __attribute__((ext_vector_type(8)))  float  v8f;
typedef __attribute__((ext_vector_type(4)))  float  v4f;

#define BB 1024
#define SS 256
#define DD 128
#define HH 256

// scheduler grouping: 0x008=WMMA, 0x020=VMEM read, 0x100=DS read
#if __has_builtin(__builtin_amdgcn_sched_group_barrier)
#define SCHED_GROUP(mask, size) __builtin_amdgcn_sched_group_barrier((mask), (size), 0)
#else
#define SCHED_GROUP(mask, size)
#endif

// ---------- helpers ----------

// native fptrunc float->bf16 (RNE; selects v_cvt_pk_bf16_f32 on gfx1250)
__device__ __forceinline__ bf16_t to_bf(float x) { return (bf16_t)x; }

__device__ __forceinline__ v4bf to_bf4(v4f x) {
  v4bf r;
#pragma unroll
  for (int j = 0; j < 4; ++j) r[j] = to_bf(x[j]);
  return r;
}

__device__ __forceinline__ v8f zero8() {
  v8f z;
#pragma unroll
  for (int i = 0; i < 8; ++i) z[i] = 0.0f;
  return z;
}

// branch-free transcendentals (v_exp_f32 / v_rcp_f32 / v_tanh_f32)
__device__ __forceinline__ float fexp2(float x) {
#if __has_builtin(__builtin_amdgcn_exp2f)
  return __builtin_amdgcn_exp2f(x);
#else
  return exp2f(x);
#endif
}
__device__ __forceinline__ float frcp(float x) {
#if __has_builtin(__builtin_amdgcn_rcpf)
  return __builtin_amdgcn_rcpf(x);
#else
  return 1.0f / x;
#endif
}
#define LOG2E 1.4426950408889634f
__device__ __forceinline__ float fsigmoid(float x) {
  return frcp(1.0f + fexp2(-LOG2E * x));
}
__device__ __forceinline__ float ftanh(float x) {
#if __has_builtin(__builtin_amdgcn_tanhf)
  return __builtin_amdgcn_tanhf(x);
#elif __has_builtin(__builtin_amdgcn_tanh_f32)
  return __builtin_amdgcn_tanh_f32(x);
#else
  return 2.0f * frcp(1.0f + fexp2(-2.0f * LOG2E * x)) - 1.0f;  // 2*sigmoid(2x)-1
#endif
}

// B-fragment: packed layout, lane reads 16 contiguous bf16 (32B) -> 2x global_load_b128
__device__ __forceinline__ v16bf load_bfrag(const bf16_t* __restrict__ base, int block, int lane) {
  const bf16_t* p = base + ((size_t)block * 32 + lane) * 16;
  v16bf r;
#pragma unroll
  for (int j = 0; j < 16; ++j) r[j] = p[j];
  return r;
}

// A-fragment from LDS row-major [16][stride] bf16 (contiguous k = 16*kh + j, matches packed B)
__device__ __forceinline__ v16bf load_afrag(const bf16_t* __restrict__ A, int stride,
                                            int m, int kh, int k0) {
  const bf16_t* p0 = A + m * stride + k0 + kh * 16;
  v16bf r;
#pragma unroll
  for (int j = 0; j < 8; ++j) { r[j] = p0[j]; r[j + 8] = p0[8 + j]; }
  return r;
}

__device__ __forceinline__ v8f wmma_bf16(v16bf a, v16bf b, v8f c) {
  return __builtin_amdgcn_wmma_f32_16x16x32_bf16(false, a, false, b, (short)0, c, false, false);
}

// ---------- workspace layout (bytes) ----------
#define WGP_OFF 0        // bf16 48*16*512 = 393216 elems (786432 B)
#define W1P_OFF 786432   // bf16  4* 8*512 =  16384 elems ( 32768 B)
#define W2P_OFF 819200   // bf16 16* 2*512 =  16384 elems ( 32768 B)
#define BG_OFF  851968   // f32 1024 (4096 B)
#define ADT_OFF 856064   // f32  256 (1024 B)

// ---------- prep: pack weights into WMMA B-fragment layout ----------
__global__ void odernn_pack_kernel(const float* __restrict__ W1, const float* __restrict__ W2,
                                   const float* __restrict__ w_ih, const float* __restrict__ w_hh,
                                   const float* __restrict__ b_ih, const float* __restrict__ b_hh,
                                   bf16_t* __restrict__ W1p, bf16_t* __restrict__ W2p,
                                   bf16_t* __restrict__ Wgp, float* __restrict__ bg) {
  int idx = blockIdx.x * 256 + threadIdx.x;
  if (idx < 393216) {   // Wg[n][k]: k<256 -> w_ih[n][k], else w_hh[n][k-256]
    int j = idx & 15, lane = (idx >> 4) & 31, blk = idx >> 9;
    int kt = blk & 15, nt = blk >> 4;
    int k = kt * 32 + (lane >> 4) * 16 + j;
    int n = nt * 16 + (lane & 15);
    float v = (k < 256) ? w_ih[(size_t)n * 256 + k] : w_hh[(size_t)n * 256 + (k - 256)];
    Wgp[idx] = to_bf(v);
  }
  if (idx < 16384) {
    int j = idx & 15, lane = (idx >> 4) & 31, blk = idx >> 9;
    {   // W1 [256][64]: blocks nt*8+kt
      int kt = blk & 7, nt = blk >> 3;
      int k = kt * 32 + (lane >> 4) * 16 + j;
      int n = nt * 16 + (lane & 15);
      W1p[idx] = to_bf(W1[k * 64 + n]);
    }
    {   // W2 [64][256]: blocks nt*2+kt
      int kt = blk & 1, nt = blk >> 1;
      int k = kt * 32 + (lane >> 4) * 16 + j;
      int n = nt * 16 + (lane & 15);
      W2p[idx] = to_bf(W2[k * 256 + n]);
    }
  }
  if (idx < 1024) {
    float v = (idx < 512) ? (b_ih[idx] + b_hh[idx])
            : (idx < 768) ? b_ih[idx]
                          : b_hh[idx - 256];
    bg[idx] = v;
  }
}

// ---------- prep: batch-averaged time deltas ----------
__global__ void odernn_avgdt_kernel(const float* __restrict__ times, float* __restrict__ avgdt) {
  int s = blockIdx.x;
  __shared__ float red[256];
  float acc = 0.0f;
  if (s > 0)
    for (int b = threadIdx.x; b < BB; b += 256)
      acc += times[(size_t)b * SS + s] - times[(size_t)b * SS + s - 1];
  red[threadIdx.x] = acc;
  __syncthreads();
  for (int off = 128; off > 0; off >>= 1) {
    if (threadIdx.x < off) red[threadIdx.x] += red[threadIdx.x + off];
    __syncthreads();
  }
  if (threadIdx.x == 0) avgdt[s] = red[0] * (1.0f / (float)BB);
}

// ---------- main persistent ODE-RNN kernel ----------
// grid = 64 (16-row batch tiles), block = 128 (4 wave32), dyn LDS = 131072 B
// launch_bounds(128, 2): cap VGPRs so 2 workgroups co-reside per WGP.
__global__ __launch_bounds__(128, 2)
void odernn_scan_kernel(const float* __restrict__ vals, const float* __restrict__ msks,
                        const int* __restrict__ lengths, const float* __restrict__ h0,
                        const float* __restrict__ b1, const float* __restrict__ b2,
                        const bf16_t* __restrict__ W1p, const bf16_t* __restrict__ W2p,
                        const bf16_t* __restrict__ Wgp, const float* __restrict__ bg,
                        const float* __restrict__ avgdt, float* __restrict__ out) {
  const float BT[5][5] = {
      {0.2f, 0.f, 0.f, 0.f, 0.f},
      {3.f / 40.f, 9.f / 40.f, 0.f, 0.f, 0.f},
      {44.f / 45.f, -56.f / 15.f, 32.f / 9.f, 0.f, 0.f},
      {19372.f / 6561.f, -25360.f / 2187.f, 64448.f / 6561.f, -212.f / 729.f, 0.f},
      {9017.f / 3168.f, -355.f / 33.f, 46732.f / 5247.f, 49.f / 176.f, -5103.f / 18656.f}};
  const float B5[6] = {35.f / 384.f, 0.f, 500.f / 1113.f, 125.f / 192.f,
                       -2187.f / 6784.f, 11.f / 84.f};

  const int tid = threadIdx.x;
  const int w = tid >> 5;           // wave id 0..3
  const int lane = tid & 31;
  const int ln = lane & 15;         // N-column / A-row within tile
  const int kh = lane >> 4;         // A/B lane-half (K partition)
  const int mh = lane >> 4;         // C/D lane-half (M partition)
  const int m0 = blockIdx.x * 16;   // batch row base

  extern __shared__ char smem[];
  float* ksg = (float*)smem;                 // 6*16*256 f32 (RK stages) / 16*1024 f32 (GRU gates)
  float* hsh = (float*)(smem + 98304);       // 16*256 f32 hidden state
  bf16_t* Ab = (bf16_t*)(smem + 114688);     // 16*512 bf16 A-operand staging

  // --- persistent register-resident W1/W2 B-fragments ---
  v16bf w1f[8];
#pragma unroll
  for (int kt = 0; kt < 8; ++kt) w1f[kt] = load_bfrag(W1p, w * 8 + kt, lane);
  v16bf w2f[4][2];
#pragma unroll
  for (int q = 0; q < 4; ++q)
#pragma unroll
    for (int kt = 0; kt < 2; ++kt) w2f[q][kt] = load_bfrag(W2p, (w * 4 + q) * 2 + kt, lane);

  const float b1v = b1[w * 16 + ln];
  float b2v[4];
#pragma unroll
  for (int q = 0; q < 4; ++q) b2v[q] = b2[(w * 4 + q) * 16 + ln];

  // --- init hidden state (float4) ---
  for (int q4 = tid; q4 < 1024; q4 += 128) {
    int m = q4 >> 6, c4 = (q4 & 63) * 4;
    *(v4f*)&hsh[m * 256 + c4] = *(const v4f*)&h0[c4];
  }
  __syncthreads();

  for (int s = 0; s < SS; ++s) {
    const float dt = avgdt[s];      // uniform across grid -> uniform branch
    if (dt > 0.0f) {
      const float hstep = dt * 0.25f;   // NSTEPS = 4
      for (int rk = 0; rk < 4; ++rk) {
#pragma unroll
        for (int j = 0; j < 6; ++j) {
          // stage arg: harg = h + hstep * sum_p BT[j-1][p] * k_p  (float4 -> 4x bf16)
          for (int q4 = tid; q4 < 1024; q4 += 128) {
            int m = q4 >> 6, c4 = (q4 & 63) * 4;
            v4f v = *(const v4f*)&hsh[m * 256 + c4];
            if (j > 0) {
              const float* kb = ksg + m * 256 + c4;
              v4f a = (*(const v4f*)(kb)) * BT[j - 1][0];
#pragma unroll
              for (int p = 1; p < j; ++p)
                a += (*(const v4f*)(kb + p * 4096)) * BT[j - 1][p];
              v += a * hstep;
            }
            *(v4bf*)&Ab[m * 512 + c4] = to_bf4(v);
          }
          __syncthreads();
          // matmul1: [16x256]@[256x64] ; all 16 DS reads first, then 8 WMMAs
          v16bf af1[8];
#pragma unroll
          for (int kt = 0; kt < 8; ++kt) af1[kt] = load_afrag(Ab, 512, ln, kh, kt * 32);
          v8f acc = zero8();
#pragma unroll
          for (int kt = 0; kt < 8; ++kt) acc = wmma_bf16(af1[kt], w1f[kt], acc);
          SCHED_GROUP(0x100, 16);   // 16x ds_load_b128 (A fragments)
          SCHED_GROUP(0x008, 8);    // 8x v_wmma back-to-back
#pragma unroll
          for (int r = 0; r < 8; ++r)
            Ab[(mh * 8 + r) * 512 + 256 + w * 16 + ln] = to_bf(ftanh(acc[r] + b1v));
          __syncthreads();
          // matmul2: [16x64]@[64x256] -> k_j ; A-frags shared across all 4 N-tiles
          v16bf af2[2];
          af2[0] = load_afrag(Ab, 512, ln, kh, 256);
          af2[1] = load_afrag(Ab, 512, ln, kh, 288);
#pragma unroll
          for (int q = 0; q < 4; ++q) {
            v8f a2 = zero8();
            a2 = wmma_bf16(af2[0], w2f[q][0], a2);
            a2 = wmma_bf16(af2[1], w2f[q][1], a2);
            int nc = (w * 4 + q) * 16 + ln;
#pragma unroll
            for (int r = 0; r < 8; ++r)
              ksg[(j * 16 + mh * 8 + r) * 256 + nc] = a2[r] + b2v[q];
          }
          SCHED_GROUP(0x100, 4);    // 4x ds_load_b128 (A fragments)
          SCHED_GROUP(0x008, 8);    // 8x v_wmma
          __syncthreads();
        }
        // h += hstep * (b1*k1 + b3*k3 + b4*k4 + b5*k5 + b6*k6)   (float4)
        for (int q4 = tid; q4 < 1024; q4 += 128) {
          int m = q4 >> 6, c4 = (q4 & 63) * 4;
          const float* kb = ksg + m * 256 + c4;
          v4f a = (*(const v4f*)(kb + 0 * 4096)) * B5[0]
                + (*(const v4f*)(kb + 2 * 4096)) * B5[2]
                + (*(const v4f*)(kb + 3 * 4096)) * B5[3]
                + (*(const v4f*)(kb + 4 * 4096)) * B5[4]
                + (*(const v4f*)(kb + 5 * 4096)) * B5[5];
          v4f* hp = (v4f*)&hsh[m * 256 + c4];
          *hp = *hp + a * hstep;
        }
        __syncthreads();
      }
    }
    // --- GRU: build A = [val | msk | h] (16 x 512 bf16, float4 path) ---
    for (int q4 = tid; q4 < 2048; q4 += 128) {
      int m = q4 >> 7, c4 = (q4 & 127) * 4;
      v4f v;
      if (c4 < 128)      v = *(const v4f*)&vals[((size_t)(m0 + m) * SS + s) * DD + c4];
      else if (c4 < 256) v = *(const v4f*)&msks[((size_t)(m0 + m) * SS + s) * DD + (c4 - 128)];
      else               v = *(const v4f*)&hsh[m * 256 + (c4 - 256)];
      *(v4bf*)&Ab[m * 512 + c4] = to_bf4(v);
    }
    // prefetch next timestep's inputs into cache (global_prefetch_b8)
    if (s + 1 < SS) {
      int rowp = tid >> 3, seg = tid & 7;
      const float* p = (seg < 4 ? vals : msks) +
                       ((size_t)(m0 + rowp) * SS + (s + 1)) * DD + (seg & 3) * 32;
      __builtin_prefetch(p, 0, 1);
    }
    __syncthreads();
    // --- GRU gate GEMMs: per-wave A-frags loaded ONCE, reused by all jobs ---
    v16bf ag[16];
#pragma unroll
    for (int kt = 0; kt < 16; ++kt) ag[kt] = load_afrag(Ab, 512, ln, kh, kt * 32);
    // class 1: fused r/z gates, 8 N-tiles per wave, K = 512 (software-pipelined B loads)
#pragma unroll
    for (int q = 0; q < 8; ++q) {
      const int pnt = w * 8 + q;
      const int ob = pnt * 16;
      v8f acc = zero8();
      v16bf bcur = load_bfrag(Wgp, pnt * 16, lane);
#pragma unroll
      for (int kt = 0; kt < 16; ++kt) {
        v16bf bn = (kt + 1 < 16) ? load_bfrag(Wgp, pnt * 16 + kt + 1, lane) : bcur;
        acc = wmma_bf16(ag[kt], bcur, acc);
        bcur = bn;
      }
      float bb = bg[ob + ln];
#pragma unroll
      for (int r = 0; r < 8; ++r)
        ksg[(mh * 8 + r) * 1024 + ob + ln] = acc[r] + bb;
    }
    // class 2+3: n-gate i_n (K-tiles 0..7) and h_n (K-tiles 8..15), 4 N-tiles per wave
#pragma unroll
    for (int q = 0; q < 4; ++q) {
      const int pnt = 32 + w * 4 + q;
      {   // i_n
        v8f acc = zero8();
        v16bf bcur = load_bfrag(Wgp, pnt * 16, lane);
#pragma unroll
        for (int kt = 0; kt < 8; ++kt) {
          v16bf bn = (kt + 1 < 8) ? load_bfrag(Wgp, pnt * 16 + kt + 1, lane) : bcur;
          acc = wmma_bf16(ag[kt], bcur, acc);
          bcur = bn;
        }
        int ob = 512 + (pnt - 32) * 16;
        float bb = bg[ob + ln];
#pragma unroll
        for (int r = 0; r < 8; ++r)
          ksg[(mh * 8 + r) * 1024 + ob + ln] = acc[r] + bb;
      }
      {   // h_n
        v8f acc = zero8();
        v16bf bcur = load_bfrag(Wgp, pnt * 16 + 8, lane);
#pragma unroll
        for (int kt = 8; kt < 16; ++kt) {
          v16bf bn = (kt + 1 < 16) ? load_bfrag(Wgp, pnt * 16 + kt + 1, lane) : bcur;
          acc = wmma_bf16(ag[kt], bcur, acc);
          bcur = bn;
        }
        int ob = 768 + (pnt - 32) * 16;
        float bb = bg[ob + ln];
#pragma unroll
        for (int r = 0; r < 8; ++r)
          ksg[(mh * 8 + r) * 1024 + ob + ln] = acc[r] + bb;
      }
    }
    // schedule shape: 32 A-frag DS reads up front, then B-loads one group
    // ahead of their consuming WMMAs (8 vmem reads : 4 wmma), 48 repeats
    SCHED_GROUP(0x100, 32);
#pragma unroll
    for (int it = 0; it < 48; ++it) {
      SCHED_GROUP(0x020, 8);
      SCHED_GROUP(0x008, 4);
    }
    __syncthreads();
    // --- GRU elementwise update (float4; invalid rows keep post-ODE h) ---
    for (int q4 = tid; q4 < 1024; q4 += 128) {
      int m = q4 >> 6, c4 = (q4 & 63) * 4;
      const float* g = ksg + m * 1024;
      v4f gr = *(const v4f*)(g + c4);
      v4f gz = *(const v4f*)(g + 256 + c4);
      v4f gn = *(const v4f*)(g + 512 + c4);
      v4f gh = *(const v4f*)(g + 768 + c4);
      v4f hold = *(const v4f*)&hsh[m * 256 + c4];
      v4f hn;
#pragma unroll
      for (int u = 0; u < 4; ++u) {
        float r = fsigmoid(gr[u]);
        float z = fsigmoid(gz[u]);
        float nn = ftanh(gn[u] + r * gh[u]);
        hn[u] = (1.0f - z) * nn + z * hold[u];
      }
      bool valid = s < lengths[m0 + m];
      *(v4f*)&hsh[m * 256 + c4] = valid ? hn : hold;
    }
    __syncthreads();
  }
  // --- write final h [B,H] fp32 (float4) ---
  for (int q4 = tid; q4 < 1024; q4 += 128) {
    int m = q4 >> 6, c4 = (q4 & 63) * 4;
    *(v4f*)&out[(size_t)(m0 + m) * 256 + c4] = *(const v4f*)&hsh[m * 256 + c4];
  }
}

extern "C" void kernel_launch(void* const* d_in, const int* in_sizes, int n_in,
                              void* d_out, int out_size, void* d_ws, size_t ws_size,
                              hipStream_t stream) {
  const float* batch_times  = (const float*)d_in[0];
  const float* batch_values = (const float*)d_in[1];
  const float* batch_masks  = (const float*)d_in[2];
  const int*   lengths      = (const int*)d_in[3];
  const float* h0           = (const float*)d_in[4];
  const float* W1           = (const float*)d_in[5];
  const float* b1           = (const float*)d_in[6];
  const float* W2           = (const float*)d_in[7];
  const float* b2           = (const float*)d_in[8];
  const float* w_ih         = (const float*)d_in[9];
  const float* w_hh         = (const float*)d_in[10];
  const float* b_ih         = (const float*)d_in[11];
  const float* b_hh         = (const float*)d_in[12];

  char* ws = (char*)d_ws;
  bf16_t* Wgp  = (bf16_t*)(ws + WGP_OFF);
  bf16_t* W1p  = (bf16_t*)(ws + W1P_OFF);
  bf16_t* W2p  = (bf16_t*)(ws + W2P_OFF);
  float*  bg   = (float*)(ws + BG_OFF);
  float*  adt  = (float*)(ws + ADT_OFF);

  odernn_pack_kernel<<<1536, 256, 0, stream>>>(W1, W2, w_ih, w_hh, b_ih, b_hh,
                                               W1p, W2p, Wgp, bg);
  odernn_avgdt_kernel<<<SS, 256, 0, stream>>>(batch_times, adt);
  odernn_scan_kernel<<<BB / 16, 128, 131072, stream>>>(
      batch_values, batch_masks, lengths, h0, b1, b2, W1p, W2p, Wgp, bg, adt,
      (float*)d_out);
}
